// Model_RWKV7_10754598109497
// MI455X (gfx1250) — matrix operand order repaired and checked by reading, not compiled, not run on hardware
//
#include <hip/hip_runtime.h>
#include <hip/hip_bf16.h>
#include <cstdint>

typedef __attribute__((ext_vector_type(16))) _Float16 v16h;
typedef __attribute__((ext_vector_type(8)))  _Float16 v8h;
typedef __attribute__((ext_vector_type(8)))  float    v8f;

#define BM 128
#define BN 128
#define BK 32
#define BKP 40   // padded LDS row (halves) -> conflict-free b128 fragment gathers

// ---------------------------------------------------------------------------
// WMMA GEMM:  C[M,Nd] (f32) = epi( A[M,Kd] (f32, optional token-shift lerp) @ Bt^T )
// Bt is pre-transposed f16 weights laid out [Nd][Kd].
// EPI: 0 store, 1 exp(-exp(x+bias)), 2 sigmoid(x+bias), 3 sigmoid(x),
//      4 accumulate into C, 5 squared-relu store.
// LERP: apply x + (shift(x) - x) * mu[col] while loading A (branchless).
// Double-buffered LDS; A staged via registers (lerp+cvt is VALU work anyway),
// B copied with gfx1250 async global->LDS (no VGPR staging, ASYNCcnt-tracked).
// ---------------------------------------------------------------------------
template<int EPI, bool LERP>
__global__ void __launch_bounds__(256, 1) gemm_wmma_kernel(
    const float* __restrict__ A, const _Float16* __restrict__ Bt,
    float* __restrict__ C, const float* __restrict__ mu,
    const float* __restrict__ bias, int M, int Kd, int Nd, int T)
{
    __shared__ __align__(16) _Float16 sA[2][BM][BKP];
    __shared__ __align__(16) _Float16 sB[2][BN][BKP];

    const int tid  = threadIdx.x;
    const int lane = tid & 31;
    const int wid  = tid >> 5;
    const int wm   = wid & 3;   // 4 waves along M (32 rows each)
    const int wn   = wid >> 2;  // 2 waves along N (64 cols each)
    const int rowBase = blockIdx.y * BM;
    const int colBase = blockIdx.x * BN;

    v8f acc[2][4];
#pragma unroll
    for (int i = 0; i < 2; ++i)
#pragma unroll
        for (int j = 0; j < 4; ++j) acc[i][j] = (v8f){};

    const int lrow  = tid >> 1;          // 0..127 (tile row this thread loads)
    const int lkoff = (tid & 1) * 16;    // 0 or 16 (16 halves per thread)

    // loop-invariant load pointers (branchless token shift)
    const int    grow = rowBase + lrow;
    const bool   t0   = ((grow % T) == 0);
    const float* arow = A + (size_t)grow * Kd + lkoff;
    const float* prow = t0 ? arow : (arow - Kd);    // always in-bounds
    const float  ps   = t0 ? 0.0f : 1.0f;           // zeroes the shifted term at t==0
    const _Float16* brow = Bt + (size_t)(colBase + lrow) * Kd + lkoff;

    float a16[16];

    // ---- A staging: global f32 -> regs (lerp) ----
    auto load_a_regs = [&](int kb) {
#pragma unroll
        for (int q = 0; q < 4; ++q) {
            float4 t = *(const float4*)(arow + kb + 4 * q);
            a16[4*q+0] = t.x; a16[4*q+1] = t.y; a16[4*q+2] = t.z; a16[4*q+3] = t.w;
        }
        if (LERP) {
#pragma unroll
            for (int q = 0; q < 4; ++q) {
                float4 p = *(const float4*)(prow + kb + 4 * q);
                float4 m = *(const float4*)(mu + kb + lkoff + 4 * q);
                a16[4*q+0] += (p.x * ps - a16[4*q+0]) * m.x;
                a16[4*q+1] += (p.y * ps - a16[4*q+1]) * m.y;
                a16[4*q+2] += (p.z * ps - a16[4*q+2]) * m.z;
                a16[4*q+3] += (p.w * ps - a16[4*q+3]) * m.w;
            }
        }
    };
    auto store_a_lds = [&](int buf) {
        v8h h0, h1;
#pragma unroll
        for (int j = 0; j < 8; ++j) { h0[j] = (_Float16)a16[j]; h1[j] = (_Float16)a16[j + 8]; }
        *(v8h*)&sA[buf][lrow][lkoff]     = h0;
        *(v8h*)&sA[buf][lrow][lkoff + 8] = h1;
    };
    // ---- B tile: gfx1250 async global->LDS copy, no VGPR staging ----
    auto issue_b_async = [&](int kb, int buf) {
        uint32_t dsoff = (uint32_t)(uintptr_t)(void*)&sB[buf][lrow][lkoff];
        uint64_t ga    = (uint64_t)(uintptr_t)(brow + kb);
        asm volatile(
            "global_load_async_to_lds_b128 %0, %1, off\n\t"
            "global_load_async_to_lds_b128 %2, %3, off"
            :: "v"(dsoff), "v"(ga), "v"(dsoff + 16u), "v"(ga + 16ull)
            : "memory");
    };

    // prologue: stage tile 0 into buffer 0
    issue_b_async(0, 0);
    load_a_regs(0);
    store_a_lds(0);

    for (int kb = 0; kb < Kd; kb += BK) {
        const int cur = (kb / BK) & 1;
        asm volatile("s_wait_asynccnt 0" ::: "memory");  // our B-tile writes landed
        __syncthreads();                                 // tile[cur] visible to all

        const bool more = (kb + BK) < Kd;
        if (more) {
            issue_b_async(kb + BK, cur ^ 1);   // async copy runs under the WMMAs
            load_a_regs(kb + BK);              // A loads in flight over the WMMAs
        }

        // ---- fragments per ISA VGPR layouts + WMMAs ----
        v16h af[2];
#pragma unroll
        for (int mt = 0; mt < 2; ++mt) {
            const int r   = wm * 32 + mt * 16 + (lane & 15);
            const int off = (lane >= 16) ? 8 : 0;   // A: K 0-7/16-23 vs 8-15/24-31
            v8h lo = *(const v8h*)&sA[cur][r][off];
            v8h hi = *(const v8h*)&sA[cur][r][off + 16];
#pragma unroll
            for (int j = 0; j < 8; ++j) { af[mt][j] = lo[j]; af[mt][j + 8] = hi[j]; }
        }
#pragma unroll
        for (int nt = 0; nt < 4; ++nt) {
            const int n    = wn * 64 + nt * 16 + (lane & 15);
            const int boff = (lane >= 16) ? 8 : 0;  // B: K 0-7/16-23 vs 8-15/24-31
            v8h b0 = *(const v8h*)&sB[cur][n][boff];
            v8h b1 = *(const v8h*)&sB[cur][n][boff + 16];
            v16h bf;
#pragma unroll
            for (int j = 0; j < 8; ++j) { bf[j] = b0[j]; bf[j + 8] = b1[j]; }
#pragma unroll
            for (int mt = 0; mt < 2; ++mt) {
                acc[mt][nt] = __builtin_amdgcn_wmma_f32_16x16x32_f16(
                    false, af[mt], false, bf, (short)0, acc[mt][nt], false, false);
            }
        }

        if (more) store_a_lds(cur ^ 1);        // fill the other A buffer
    }

    // ---- epilogue (C 16x16 f32 layout: VGPR i -> M=i (+8 upper half-wave)) ----
#pragma unroll
    for (int mt = 0; mt < 2; ++mt)
#pragma unroll
        for (int nt = 0; nt < 4; ++nt)
#pragma unroll
            for (int i = 0; i < 8; ++i) {
                const int row = rowBase + wm * 32 + mt * 16 + i + ((lane >= 16) ? 8 : 0);
                const int col = colBase + wn * 64 + nt * 16 + (lane & 15);
                float v = acc[mt][nt][i];
                const size_t idx = (size_t)row * Nd + col;
                if (EPI == 0)      C[idx] = v;
                else if (EPI == 1) C[idx] = __expf(-__expf(v + bias[col]));
                else if (EPI == 2) C[idx] = 1.0f / (1.0f + __expf(-(v + bias[col])));
                else if (EPI == 3) C[idx] = 1.0f / (1.0f + __expf(-v));
                else if (EPI == 4) C[idx] += v;
                else { float rl = fmaxf(v, 0.0f); C[idx] = rl * rl; }
            }
}

// ---------------------------------------------------------------------------
// Weight transpose + f32->f16:  W[Kd][Nd] -> Wt[Nd][Kd]
// ---------------------------------------------------------------------------
__global__ void transpose_to_f16_kernel(const float* __restrict__ W,
                                        _Float16* __restrict__ Wt, int Kd, int Nd)
{
    size_t idx = (size_t)blockIdx.x * 256 + threadIdx.x;
    if (idx >= (size_t)Kd * Nd) return;
    size_t k = idx / Nd, n = idx % Nd;
    Wt[n * (size_t)Kd + k] = (_Float16)W[idx];
}

// ---------------------------------------------------------------------------
// Patch embedding + sinusoidal positional encoding
// ---------------------------------------------------------------------------
__global__ void __launch_bounds__(256) patch_embed_kernel(
    const float* __restrict__ x, const float* __restrict__ Wp,
    const float* __restrict__ bp, float* __restrict__ Hout)
{
    __shared__ float pin[16];
    const int row = blockIdx.x;      // n*T + t
    const int t = row & 63;
    const int n = row >> 6;
    const int tid = threadIdx.x;
    if (tid < 16) pin[tid] = x[(size_t)n * 1024 + t * 16 + tid];
    __syncthreads();
    for (int d = tid; d < 768; d += 256) {
        float acc = bp[d];
#pragma unroll
        for (int p = 0; p < 16; ++p) acc += pin[p] * Wp[p * 768 + d];
        const int i2 = d & ~1;
        const float freq = __expf((float)i2 * (-9.210340371976184f / 768.0f));
        const float ang = (float)t * freq;
        acc += (d & 1) ? __cosf(ang) : __sinf(ang);
        Hout[(size_t)row * 768 + d] = acc;
    }
}

// ---------------------------------------------------------------------------
// Row LayerNorm (eps = 1e-5), f32 -> f32
// ---------------------------------------------------------------------------
__global__ void __launch_bounds__(256) layernorm_kernel(
    const float* __restrict__ X, const float* __restrict__ g,
    const float* __restrict__ b, float* __restrict__ Y, int D)
{
    __shared__ float red[256];
    __shared__ float red2[256];
    const int row = blockIdx.x;
    const int tid = threadIdx.x;
    const float* xr = X + (size_t)row * D;
    float s = 0.f, ss = 0.f;
    for (int d = tid; d < D; d += 256) { float v = xr[d]; s += v; ss += v * v; }
    red[tid] = s; red2[tid] = ss; __syncthreads();
    for (int st = 128; st > 0; st >>= 1) {
        if (tid < st) { red[tid] += red[tid + st]; red2[tid] += red2[tid + st]; }
        __syncthreads();
    }
    const float m = red[0] / D;
    const float var = red2[0] / D - m * m;
    const float rstd = rsqrtf(var + 1e-5f);
    float* yr = Y + (size_t)row * D;
    for (int d = tid; d < D; d += 256) yr[d] = (xr[d] - m) * rstd * g[d] + b[d];
}

// ---------------------------------------------------------------------------
// RWKV7 generalized delta-rule scan + per-head GroupNorm + gate
// one 64-thread block per (n,h); thread = v-row; S[v][*] in 64 VGPRs
// ---------------------------------------------------------------------------
__global__ void __launch_bounds__(64) rwkv7_kernel(
    const float* __restrict__ rB, const float* __restrict__ wB,
    const float* __restrict__ kB, const float* __restrict__ vB,
    const float* __restrict__ aB, const float* __restrict__ gB,
    const float* __restrict__ gn_g, const float* __restrict__ gn_b,
    float* __restrict__ yB)
{
    const int nh = blockIdx.x;           // n*H + h
    const int n = nh / 12, h = nh % 12;
    const int tid = threadIdx.x;         // v index 0..63
    const int cb = h * 64;
    __shared__ float ks[64], ws[64], as_[64], rs[64], ys[64];
    float S[64];
#pragma unroll
    for (int i = 0; i < 64; ++i) S[i] = 0.f;

    for (int t = 0; t < 64; ++t) {
        const size_t idx = ((size_t)(n * 64 + t)) * 768 + cb + tid;
        ks[tid] = kB[idx]; ws[tid] = wB[idx]; as_[tid] = aB[idx]; rs[tid] = rB[idx];
        const float vval = vB[idx];
        __syncthreads();

        float ssq = 0.f;
#pragma unroll
        for (int i = 0; i < 64; ++i) ssq += ks[i] * ks[i];
        const float rn = 1.0f / (sqrtf(ssq) + 1e-6f);   // kk = k * rn

        float sa = 0.f;
#pragma unroll
        for (int i = 0; i < 64; ++i) { float si = S[i] * ws[i]; S[i] = si; sa += si * ks[i]; }
        sa *= rn;                                       // sa = (S*w) . kk
#pragma unroll
        for (int i = 0; i < 64; ++i)
            S[i] += vval * ks[i] - sa * rn * ks[i] * as_[i];   // -outer(sa, kk*a) + outer(v,k)

        float y = 0.f;
#pragma unroll
        for (int i = 0; i < 64; ++i) y += S[i] * rs[i];

        ys[tid] = y; __syncthreads();
        float s1 = 0.f, s2 = 0.f;
#pragma unroll
        for (int i = 0; i < 64; ++i) { float yy = ys[i]; s1 += yy; s2 += yy * yy; }
        const float m = s1 * (1.0f / 64.0f);
        const float var = s2 * (1.0f / 64.0f) - m * m;
        float o = (y - m) * rsqrtf(var + 64e-5f);
        o = o * gn_g[cb + tid] + gn_b[cb + tid];
        yB[idx] = o * gB[idx];                          // fused output gate
    }
}

// ---------------------------------------------------------------------------
// Final projection: out[row, 0..15] = h[row] @ W_proj + b_proj
// ---------------------------------------------------------------------------
__global__ void __launch_bounds__(256) proj_kernel(
    const float* __restrict__ Hin, const float* __restrict__ Wp,
    const float* __restrict__ bp, float* __restrict__ out)
{
    __shared__ float sh[768];
    __shared__ float part[256];
    const int row = blockIdx.x;
    const int tid = threadIdx.x;
    const float* hr = Hin + (size_t)row * 768;
    for (int d = tid; d < 768; d += 256) sh[d] = hr[d];
    __syncthreads();
    const int p = tid & 15;
    const int c = tid >> 4;                 // 16 chunks of 48
    float s = 0.f;
    for (int d = c * 48; d < c * 48 + 48; ++d) s += sh[d] * Wp[d * 16 + p];
    part[tid] = s; __syncthreads();
    if (tid < 16) {
        float tot = bp[tid];
#pragma unroll
        for (int c2 = 0; c2 < 16; ++c2) tot += part[c2 * 16 + tid];
        out[(size_t)row * 16 + tid] = tot;
    }
}

// ---------------------------------------------------------------------------
extern "C" void kernel_launch(void* const* d_in, const int* in_sizes, int n_in,
                              void* d_out, int out_size, void* d_ws, size_t ws_size,
                              hipStream_t stream)
{
    const float* x       = (const float*)d_in[0];
    const float* W_patch = (const float*)d_in[1];
    const float* b_patch = (const float*)d_in[2];
    const float* ln1_g   = (const float*)d_in[3];
    const float* ln1_b   = (const float*)d_in[4];
    const float* ln2_g   = (const float*)d_in[5];
    const float* ln2_b   = (const float*)d_in[6];
    const float* mu_r    = (const float*)d_in[7];
    const float* mu_w    = (const float*)d_in[8];
    const float* mu_k    = (const float*)d_in[9];
    const float* mu_v    = (const float*)d_in[10];
    const float* mu_a    = (const float*)d_in[11];
    const float* mu_g    = (const float*)d_in[12];
    const float* mu_c    = (const float*)d_in[13];
    const float* Wr      = (const float*)d_in[14];
    const float* Wk      = (const float*)d_in[15];
    const float* Wv      = (const float*)d_in[16];
    const float* Wg      = (const float*)d_in[17];
    const float* Ww      = (const float*)d_in[18];
    const float* w0      = (const float*)d_in[19];
    const float* Wa      = (const float*)d_in[20];
    const float* a0      = (const float*)d_in[21];
    const float* Wo      = (const float*)d_in[22];
    const float* gn_g    = (const float*)d_in[23];
    const float* gn_b    = (const float*)d_in[24];
    const float* Wc1     = (const float*)d_in[25];
    const float* Wc2     = (const float*)d_in[26];
    const float* W_proj  = (const float*)d_in[27];
    const float* b_proj  = (const float*)d_in[28];

    const int M = 32768, D = 768, DFF = 3072, T = 64;

    char* base = (char*)d_ws;
    size_t off = 0;
    auto alloc = [&](size_t bytes) -> void* {
        void* p = base + off;
        off = (off + bytes + 255) & ~(size_t)255;
        return p;
    };

    float* hbuf   = (float*)alloc((size_t)M * D * 4);
    float* xxbuf  = (float*)alloc((size_t)M * D * 4);
    float* rbuf   = (float*)alloc((size_t)M * D * 4);
    float* kbuf   = (float*)alloc((size_t)M * D * 4);
    float* vbuf   = (float*)alloc((size_t)M * D * 4);
    float* wbuf   = (float*)alloc((size_t)M * D * 4);
    float* abuf   = (float*)alloc((size_t)M * D * 4);
    float* gbuf   = (float*)alloc((size_t)M * D * 4);
    float* ybuf   = (float*)alloc((size_t)M * D * 4);
    float* midbuf = (float*)alloc((size_t)M * DFF * 4);
    _Float16* WrT  = (_Float16*)alloc((size_t)D * D * 2);
    _Float16* WkT  = (_Float16*)alloc((size_t)D * D * 2);
    _Float16* WvT  = (_Float16*)alloc((size_t)D * D * 2);
    _Float16* WgT  = (_Float16*)alloc((size_t)D * D * 2);
    _Float16* WwT  = (_Float16*)alloc((size_t)D * D * 2);
    _Float16* WaT  = (_Float16*)alloc((size_t)D * D * 2);
    _Float16* WoT  = (_Float16*)alloc((size_t)D * D * 2);
    _Float16* Wc1T = (_Float16*)alloc((size_t)D * DFF * 2);
    _Float16* Wc2T = (_Float16*)alloc((size_t)DFF * D * 2);

    const int nDD = (D * D + 255) / 256;
    const int nDF = (D * DFF + 255) / 256;
    transpose_to_f16_kernel<<<nDD, 256, 0, stream>>>(Wr, WrT, D, D);
    transpose_to_f16_kernel<<<nDD, 256, 0, stream>>>(Wk, WkT, D, D);
    transpose_to_f16_kernel<<<nDD, 256, 0, stream>>>(Wv, WvT, D, D);
    transpose_to_f16_kernel<<<nDD, 256, 0, stream>>>(Wg, WgT, D, D);
    transpose_to_f16_kernel<<<nDD, 256, 0, stream>>>(Ww, WwT, D, D);
    transpose_to_f16_kernel<<<nDD, 256, 0, stream>>>(Wa, WaT, D, D);
    transpose_to_f16_kernel<<<nDD, 256, 0, stream>>>(Wo, WoT, D, D);
    transpose_to_f16_kernel<<<nDF, 256, 0, stream>>>(Wc1, Wc1T, D, DFF);
    transpose_to_f16_kernel<<<nDF, 256, 0, stream>>>(Wc2, Wc2T, DFF, D);

    patch_embed_kernel<<<M, 256, 0, stream>>>(x, W_patch, b_patch, hbuf);
    layernorm_kernel<<<M, 256, 0, stream>>>(hbuf, ln1_g, ln1_b, xxbuf, D);

    dim3 g1(D / BN, M / BM);    // (6, 256)
    gemm_wmma_kernel<0, true ><<<g1, 256, 0, stream>>>(xxbuf, WrT, rbuf, mu_r, nullptr, M, D, D, T);
    gemm_wmma_kernel<0, true ><<<g1, 256, 0, stream>>>(xxbuf, WkT, kbuf, mu_k, nullptr, M, D, D, T);
    gemm_wmma_kernel<0, true ><<<g1, 256, 0, stream>>>(xxbuf, WvT, vbuf, mu_v, nullptr, M, D, D, T);
    gemm_wmma_kernel<1, true ><<<g1, 256, 0, stream>>>(xxbuf, WwT, wbuf, mu_w, w0, M, D, D, T);
    gemm_wmma_kernel<2, true ><<<g1, 256, 0, stream>>>(xxbuf, WaT, abuf, mu_a, a0, M, D, D, T);
    gemm_wmma_kernel<3, true ><<<g1, 256, 0, stream>>>(xxbuf, WgT, gbuf, mu_g, nullptr, M, D, D, T);

    rwkv7_kernel<<<512 * 12, 64, 0, stream>>>(rbuf, wbuf, kbuf, vbuf, abuf, gbuf, gn_g, gn_b, ybuf);

    gemm_wmma_kernel<4, false><<<g1, 256, 0, stream>>>(ybuf, WoT, hbuf, nullptr, nullptr, M, D, D, T);

    layernorm_kernel<<<M, 256, 0, stream>>>(hbuf, ln2_g, ln2_b, xxbuf, D);
    dim3 g2(DFF / BN, M / BM);  // (24, 256)
    gemm_wmma_kernel<5, true ><<<g2, 256, 0, stream>>>(xxbuf, Wc1T, midbuf, mu_c, nullptr, M, D, DFF, T);
    gemm_wmma_kernel<4, false><<<g1, 256, 0, stream>>>(midbuf, Wc2T, hbuf, nullptr, nullptr, M, DFF, D, T);

    proj_kernel<<<M, 256, 0, stream>>>(hbuf, W_proj, b_proj, (float*)d_out);
}
